// RibosomeCascadeNative_75118978007116
// MI455X (gfx1250) — matrix operand order repaired and checked by reading, not compiled, not run on hardware
//
#include <hip/hip_runtime.h>
#include <math.h>

// ---------------- problem constants ----------------
#define BB 4
#define SS 2048
#define HH 1024
#define KK 16
#define NHH 16
#define DD 64
#define LL 2
#define FFD 2730
#define FFP 2752                 // FF padded to multiple of 32
#define MDEC (BB * SS)           // 8192
#define MCASC (BB * KK)          // 64
#define ATTN_SCALE 0.125f        // 1/sqrt(64)

// ---------------- CDNA5 WMMA types ----------------
typedef __attribute__((ext_vector_type(16))) __bf16 v16bf;
typedef __attribute__((ext_vector_type(8)))  float  v8f;

// ------------------------------------------------------------------
// Weight pre-conversion: fp32 (Nreal x Kd) -> bf16 (gridDim.x x Kpad),
// zero-filling row/col pads. Zero pads guarantee padded GEMMs are exact.
// ------------------------------------------------------------------
__global__ void k_cvt_w(const float* __restrict__ W, __bf16* __restrict__ O,
                        int Nreal, int Kd, int Kpad) {
  const int n = blockIdx.x;               // 0..Npad-1
  const bool vrow = (n < Nreal);
  const float* src = W + (size_t)n * (size_t)Kd;
  __bf16* dst = O + (size_t)n * (size_t)Kpad;
  for (int k = threadIdx.x * 2; k < Kpad; k += blockDim.x * 2) {
    float a = (vrow && k     < Kd) ? src[k]     : 0.0f;
    float b = (vrow && k + 1 < Kd) ? src[k + 1] : 0.0f;
    dst[k]     = (__bf16)a;
    dst[k + 1] = (__bf16)b;
  }
}

// ------------------------------------------------------------------
// A fragment: 16x32 bf16 tile of fp32 A (row-major, lda mult of 4,
// Kd mult of 32 -> always aligned float4, no guards).
// 16-bit A 16x32 (matching B): lanes 0-15 K 0..15; lanes 16-31 K 16..31.
// ------------------------------------------------------------------
__device__ __forceinline__ v16bf frag_a(const float* __restrict__ A, int lda,
                                        int grow, int k0) {
  const int lane  = threadIdx.x & 31;
  const int khalf = (lane >> 4) & 1;
  const float* p = A + (size_t)grow * (size_t)lda + k0 + khalf * 16;
  float4 q0 = *(const float4*)(p);
  float4 q1 = *(const float4*)(p + 4);
  float4 q2 = *(const float4*)(p + 8);
  float4 q3 = *(const float4*)(p + 12);
  v16bf a;
  a[0] = (__bf16)q0.x; a[1] = (__bf16)q0.y; a[2]  = (__bf16)q0.z; a[3]  = (__bf16)q0.w;
  a[4] = (__bf16)q1.x; a[5] = (__bf16)q1.y; a[6]  = (__bf16)q1.z; a[7]  = (__bf16)q1.w;
  a[8] = (__bf16)q2.x; a[9] = (__bf16)q2.y; a[10] = (__bf16)q2.z; a[11] = (__bf16)q2.w;
  a[12] = (__bf16)q3.x; a[13] = (__bf16)q3.y; a[14] = (__bf16)q3.z; a[15] = (__bf16)q3.w;
  return a;
}

// ------------------------------------------------------------------
// B fragment: 32x16 tile of B = W^T, W bf16 row-major (ldw mult of 16).
// 16-bit B 32x16 (this code's contiguous k order, same as A): lanes 0-15 K 0..15, lanes 16-31 K 16..31; N = lane&15.
// Single aligned 32-byte vector load.
// ------------------------------------------------------------------
__device__ __forceinline__ v16bf frag_b(const __bf16* __restrict__ W, int ldw,
                                        int ncol, int k0) {
  const int lane  = threadIdx.x & 31;
  const int n     = ncol + (lane & 15);
  const int kbase = ((lane >> 4) & 1) * 16;
  const __bf16* p = W + (size_t)n * (size_t)ldw + k0 + kbase;
  __builtin_prefetch(p + 32, 0, 0);      // global_prefetch next K tile
  return *(const v16bf*)(p);
}

// ------------------------------------------------------------------
// GEMM: C[M,N] = A[M,Kd](fp32) @ Wb[N,Kd](bf16)^T (+bias[n]) (+resid[m,n])
// Requirements (guaranteed by launcher): M%16==0, N%32==0, Kd%32==0.
// Block = 256 = 8 waves; wave owns one 16-row tile x 32 columns
// (two independent accumulators -> two WMMAs per K step, no hazard stall).
// ------------------------------------------------------------------
__global__ void k_gemm_bf16(const float* __restrict__ A, const __bf16* __restrict__ W,
                            const float* __restrict__ bias, const float* __restrict__ resid,
                            float* __restrict__ C, int M, int N, int Kd) {
  const int wave = threadIdx.x >> 5;
  const int lane = threadIdx.x & 31;
  const int mtile = blockIdx.y * 8 + wave;
  if (mtile * 16 >= M) return;
  const int mrow  = mtile * 16;
  const int ncol0 = blockIdx.x * 32;
  const int arow  = mrow + (lane & 15);

  v8f c0, c1;
#pragma unroll
  for (int i = 0; i < 8; ++i) { c0[i] = 0.0f; c1[i] = 0.0f; }

#pragma unroll 2
  for (int k0 = 0; k0 < Kd; k0 += 32) {
    v16bf a  = frag_a(A, Kd, arow, k0);
    v16bf b0 = frag_b(W, Kd, ncol0,      k0);
    v16bf b1 = frag_b(W, Kd, ncol0 + 16, k0);
    c0 = __builtin_amdgcn_wmma_f32_16x16x32_bf16(false, a, false, b0,
                                                 (short)0, c0, false, false);
    c1 = __builtin_amdgcn_wmma_f32_16x16x32_bf16(false, a, false, b1,
                                                 (short)0, c1, false, false);
  }

  // C layout: VGPR r -> row = mrow + r + 8*(lane>=16), col = ncol + (lane&15)
  const int r0   = mrow + ((lane >> 4) << 3);
  const int colA = ncol0 + (lane & 15);
  const int colB = colA + 16;
  const float baddA = bias ? bias[colA] : 0.0f;
  const float baddB = bias ? bias[colB] : 0.0f;
  if (resid) {
#pragma unroll
    for (int r = 0; r < 8; ++r) {
      size_t row = (size_t)(r0 + r) * (size_t)N;
      C[row + colA] = c0[r] + baddA + resid[row + colA];
      C[row + colB] = c1[r] + baddB + resid[row + colB];
    }
  } else {
#pragma unroll
    for (int r = 0; r < 8; ++r) {
      size_t row = (size_t)(r0 + r) * (size_t)N;
      C[row + colA] = c0[r] + baddA;
      C[row + colB] = c1[r] + baddB;
    }
  }
}

// ------------------------------------------------------------------
// Fused SwiGLU dual GEMM: G = silu(A@W1^T) * (A@W3^T); N%16==0, Kd%32==0.
// ------------------------------------------------------------------
__global__ void k_gemm_swiglu(const float* __restrict__ A, const __bf16* __restrict__ W1,
                              const __bf16* __restrict__ W3, float* __restrict__ G,
                              int M, int N, int Kd) {
  const int wave = threadIdx.x >> 5;
  const int lane = threadIdx.x & 31;
  const int mtile = blockIdx.y * 8 + wave;
  if (mtile * 16 >= M) return;
  const int mrow = mtile * 16;
  const int ncol = blockIdx.x * 16;
  const int arow = mrow + (lane & 15);

  v8f c1, c3;
#pragma unroll
  for (int i = 0; i < 8; ++i) { c1[i] = 0.0f; c3[i] = 0.0f; }

#pragma unroll 2
  for (int k0 = 0; k0 < Kd; k0 += 32) {
    v16bf a  = frag_a(A, Kd, arow, k0);
    v16bf b1 = frag_b(W1, Kd, ncol, k0);
    v16bf b3 = frag_b(W3, Kd, ncol, k0);
    c1 = __builtin_amdgcn_wmma_f32_16x16x32_bf16(false, a, false, b1,
                                                 (short)0, c1, false, false);
    c3 = __builtin_amdgcn_wmma_f32_16x16x32_bf16(false, a, false, b3,
                                                 (short)0, c3, false, false);
  }

  const int r0  = mrow + ((lane >> 4) << 3);
  const int col = ncol + (lane & 15);
#pragma unroll
  for (int r = 0; r < 8; ++r) {
    float x1 = c1[r];
    float g = (x1 / (1.0f + __expf(-x1))) * c3[r];   // silu(x1) * x3
    G[(size_t)(r0 + r) * (size_t)N + col] = g;
  }
}

// ------------------------------------------------------------------
// Descending stable argsort of 16 chunk weights per batch (1 wave/batch)
// ------------------------------------------------------------------
__global__ void k_sortidx(const float* __restrict__ cw, int* __restrict__ sidx) {
  const int b = blockIdx.x;
  const int lane = threadIdx.x;
  float w = (lane < KK) ? cw[b * KK + lane] : -1e30f;
  int rank = 0;
  for (int j = 0; j < KK; ++j) {
    float wj = __shfl(w, j, 32);
    if (wj > w || (wj == w && j < lane)) rank++;
  }
  if (lane < KK) sidx[b * KK + rank] = lane;
}

__global__ void k_gather(const float* __restrict__ src, const int* __restrict__ sidx,
                         float* __restrict__ dst) {
  const int b = blockIdx.x / KK, j = blockIdx.x % KK;
  const int s = sidx[b * KK + j];
  const float* in  = src + ((size_t)b * KK + s) * HH;
  float*       out = dst + ((size_t)b * KK + j) * HH;
  for (int i = threadIdx.x; i < HH; i += blockDim.x) out[i] = in[i];
}

__global__ void k_scatter(const float* __restrict__ src, const int* __restrict__ sidx,
                          float* __restrict__ dst) {
  const int b = blockIdx.x / KK, j = blockIdx.x % KK;
  const int s = sidx[b * KK + j];
  const float* in  = src + ((size_t)b * KK + j) * HH;
  float*       out = dst + ((size_t)b * KK + s) * HH;
  for (int i = threadIdx.x; i < HH; i += blockDim.x) out[i] = in[i];
}

// ------------------------------------------------------------------
// RMSNorm over last axis (one 256-thread block per row)
// ------------------------------------------------------------------
__global__ void k_rms(const float* __restrict__ in, const float* __restrict__ w,
                      float* __restrict__ out, int width) {
  const int row = blockIdx.x;
  const float* x = in + (size_t)row * width;
  float ss = 0.0f;
  for (int i = threadIdx.x; i < width; i += 256) { float v = x[i]; ss += v * v; }
#pragma unroll
  for (int o = 16; o >= 1; o >>= 1) ss += __shfl_xor(ss, o, 32);
  __shared__ float part[8];
  __shared__ float sscale;
  const int lane = threadIdx.x & 31, wave = threadIdx.x >> 5;
  if (lane == 0) part[wave] = ss;
  __syncthreads();
  if (threadIdx.x == 0) {
    float tot = 0.0f;
#pragma unroll
    for (int i = 0; i < 8; ++i) tot += part[i];
    sscale = rsqrtf(tot / (float)width + 1e-6f);
  }
  __syncthreads();
  float* y = out + (size_t)row * width;
  for (int i = threadIdx.x; i < width; i += 256) y[i] = x[i] * sscale * w[i];
}

// ------------------------------------------------------------------
// Cascade causal self-attn over K=16 sorted chunks. 1 block per (b,head).
// qkv layout: (B,K,3H) rows; q at h*64, k at H+h*64, v at 2H+h*64.
// ------------------------------------------------------------------
__global__ void k_casc_attn(const float* __restrict__ qkv, float* __restrict__ out) {
  const int b = blockIdx.x / NHH, h = blockIdx.x % NHH;
  __shared__ float sq[KK][DD], sk[KK][DD], sv[KK][DD], sc[KK][KK];
  const int t = threadIdx.x;
  for (int i = t; i < KK * DD; i += 256) {
    int kk = i >> 6, dd = i & 63;
    size_t base = ((size_t)b * KK + kk) * (3 * HH) + h * DD + dd;
    sq[kk][dd] = qkv[base];
    sk[kk][dd] = qkv[base + HH];
    sv[kk][dd] = qkv[base + 2 * HH];
  }
  __syncthreads();
  { // scores: thread t -> (i = t/16, j = t%16)
    int i = t >> 4, j = t & 15;
    float s = 0.0f;
#pragma unroll 8
    for (int dd = 0; dd < DD; ++dd) s += sq[i][dd] * sk[j][dd];
    sc[i][j] = (j > i) ? -1e30f : s * ATTN_SCALE;
  }
  __syncthreads();
  if (t < KK) { // row softmax
    float m = -1e30f;
    for (int j = 0; j < KK; ++j) m = fmaxf(m, sc[t][j]);
    float sum = 0.0f;
    for (int j = 0; j < KK; ++j) { float e = __expf(sc[t][j] - m); sc[t][j] = e; sum += e; }
    float inv = 1.0f / sum;
    for (int j = 0; j < KK; ++j) sc[t][j] *= inv;
  }
  __syncthreads();
  for (int i = t; i < KK * DD; i += 256) {
    int r = i >> 6, dd = i & 63;
    float acc = 0.0f;
#pragma unroll
    for (int j = 0; j < KK; ++j) acc += sc[r][j] * sv[j][dd];
    out[((size_t)b * KK + r) * HH + h * DD + dd] = acc;
  }
}

// ------------------------------------------------------------------
// Decoder token->chunk cross-attn. One wave per (b, h, s).
// Allowed chunks: c <= s/128 + 1 (== (s*K)//S + SLACK).
// ------------------------------------------------------------------
__global__ void k_dec_attn(const float* __restrict__ qb, const float* __restrict__ kb,
                           const float* __restrict__ vb, float* __restrict__ ob) {
  const int wid  = (blockIdx.x * blockDim.x + threadIdx.x) >> 5;
  const int lane = threadIdx.x & 31;
  const int s = wid % SS;
  const int h = (wid / SS) % NHH;
  const int b = wid / (SS * NHH);
  if (b >= BB) return;

  const int c = lane & 15;
  const int cmax = s / 128 + 1;
  const bool ok = (c <= cmax) && (c < KK);

  const float* qrow = qb + ((size_t)b * SS + s) * HH + h * DD;
  const float* krow = kb + ((size_t)b * KK + c) * HH + h * DD;
  float sc = 0.0f;
#pragma unroll
  for (int dd = 0; dd < DD; dd += 4) {
    float4 qv = *(const float4*)(qrow + dd);
    float4 kv = *(const float4*)(krow + dd);
    sc += qv.x * kv.x + qv.y * kv.y + qv.z * kv.z + qv.w * kv.w;
  }
  sc *= ATTN_SCALE;
  float e = ok ? sc : -1e30f;
  float m = e;
#pragma unroll
  for (int o = 8; o >= 1; o >>= 1) m = fmaxf(m, __shfl_xor(m, o, 32));
  float ex = ok ? __expf(e - m) : 0.0f;
  float sum = ex;
#pragma unroll
  for (int o = 8; o >= 1; o >>= 1) sum += __shfl_xor(sum, o, 32);
  float a = ex / sum;

  // output: lane owns dims {2*lane, 2*lane+1}
  float acc0 = 0.0f, acc1 = 0.0f;
  const int top = (cmax < KK - 1) ? cmax : (KK - 1);
  for (int j = 0; j <= top; ++j) {
    float aj = __shfl(a, j, 32);
    const float* vrow = vb + ((size_t)b * KK + j) * HH + h * DD + lane * 2;
    float2 vv = *(const float2*)(vrow);
    acc0 += aj * vv.x;
    acc1 += aj * vv.y;
  }
  float* orow = ob + ((size_t)b * SS + s) * HH + h * DD + lane * 2;
  float2 res; res.x = acc0; res.y = acc1;
  *(float2*)orow = res;
}

// ------------------------------------------------------------------
static inline dim3 grid_g(int M, int N) {       // k_gemm_bf16: 32 cols/block-x
  return dim3((unsigned)(N / 32), (unsigned)(((M / 16) + 7) / 8));
}
static inline dim3 grid_s(int M, int N) {       // k_gemm_swiglu: 16 cols/block-x
  return dim3((unsigned)(N / 16), (unsigned)(((M / 16) + 7) / 8));
}

extern "C" void kernel_launch(void* const* d_in, const int* in_sizes, int n_in,
                              void* d_out, int out_size, void* d_ws, size_t ws_size,
                              hipStream_t stream) {
  (void)in_sizes; (void)n_in; (void)out_size; (void)ws_size;
  const float* token   = (const float*)d_in[0];
  const float* chunk   = (const float*)d_in[1];
  const float* cweight = (const float*)d_in[2];
  const float* c_n1    = (const float*)d_in[3];
  const float* c_qkv   = (const float*)d_in[4];
  const float* c_o     = (const float*)d_in[5];
  const float* c_n2    = (const float*)d_in[6];
  const float* c_w1    = (const float*)d_in[7];
  const float* c_w2    = (const float*)d_in[8];
  const float* c_w3    = (const float*)d_in[9];
  const float* d_in_w  = (const float*)d_in[10];
  const float* d_in_b  = (const float*)d_in[11];
  const float* d_out_w = (const float*)d_in[12];
  const float* d_out_b = (const float*)d_in[13];
  const float* d_nw    = (const float*)d_in[14];
  const float* d_fnw   = (const float*)d_in[15];
  const float* d_w1    = (const float*)d_in[16];
  const float* d_w2    = (const float*)d_in[17];
  const float* d_w3    = (const float*)d_in[18];
  float* out = (float*)d_out;

  // ---- workspace carve-out (float units, 256B aligned blocks) ----
  float* ws = (float*)d_ws;
  size_t off = 0;
  auto takef = [&](size_t n) { float* p = ws + off; off += (n + 63) & ~(size_t)63; return p; };
  auto takeb = [&](size_t n) { return (__bf16*)takef(n / 2); };  // n even

  int*   sidx  = (int*)takef(MCASC);
  float* xs    = takef((size_t)MCASC * HH);
  float* hbuf  = takef((size_t)MCASC * HH);
  float* qkvb  = takef((size_t)MCASC * 3 * HH);
  float* attno = takef((size_t)MCASC * HH);
  float* gcas  = takef((size_t)MCASC * FFP);
  float* proc  = takef((size_t)MCASC * HH);
  float* qbuf  = takef((size_t)MDEC * HH);
  float* kbuf  = takef((size_t)MCASC * HH);
  float* vbuf  = takef((size_t)MCASC * HH);
  float* attn  = takef((size_t)MDEC * HH);     // attn out, later reused as rms buf
  float* t1    = takef((size_t)MDEC * HH);
  float* gated = takef((size_t)MDEC * FFP);

  // bf16 weight copies (zero-padded)
  __bf16* wb_cqkv[LL]; __bf16* wb_co[LL]; __bf16* wb_cw1[LL]; __bf16* wb_cw2[LL]; __bf16* wb_cw3[LL];
  for (int l = 0; l < LL; ++l) {
    wb_cqkv[l] = takeb((size_t)3 * HH * HH);
    wb_co[l]   = takeb((size_t)HH * HH);
    wb_cw1[l]  = takeb((size_t)FFP * HH);
    wb_cw2[l]  = takeb((size_t)HH * FFP);
    wb_cw3[l]  = takeb((size_t)FFP * HH);
  }
  __bf16* wb_inw  = takeb((size_t)3 * HH * HH);
  __bf16* wb_outw = takeb((size_t)HH * HH);
  __bf16* wb_dw1  = takeb((size_t)FFP * HH);
  __bf16* wb_dw2  = takeb((size_t)HH * FFP);
  __bf16* wb_dw3  = takeb((size_t)FFP * HH);

  // ---- convert weights to padded bf16 ----
  for (int l = 0; l < LL; ++l) {
    k_cvt_w<<<3 * HH, 256, 0, stream>>>(c_qkv + (size_t)l * 3 * HH * HH, wb_cqkv[l], 3 * HH, HH, HH);
    k_cvt_w<<<HH, 256, 0, stream>>>(c_o + (size_t)l * HH * HH, wb_co[l], HH, HH, HH);
    k_cvt_w<<<FFP, 256, 0, stream>>>(c_w1 + (size_t)l * FFD * HH, wb_cw1[l], FFD, HH, HH);
    k_cvt_w<<<FFP, 256, 0, stream>>>(c_w3 + (size_t)l * FFD * HH, wb_cw3[l], FFD, HH, HH);
    k_cvt_w<<<HH, 256, 0, stream>>>(c_w2 + (size_t)l * HH * FFD, wb_cw2[l], HH, FFD, FFP);
  }
  k_cvt_w<<<3 * HH, 256, 0, stream>>>(d_in_w, wb_inw, 3 * HH, HH, HH);
  k_cvt_w<<<HH, 256, 0, stream>>>(d_out_w, wb_outw, HH, HH, HH);
  k_cvt_w<<<FFP, 256, 0, stream>>>(d_w1, wb_dw1, FFD, HH, HH);
  k_cvt_w<<<FFP, 256, 0, stream>>>(d_w3, wb_dw3, FFD, HH, HH);
  k_cvt_w<<<HH, 256, 0, stream>>>(d_w2, wb_dw2, HH, FFD, FFP);

  // ---- CascadeProcessor ----
  k_sortidx<<<BB, 32, 0, stream>>>(cweight, sidx);
  k_gather<<<BB * KK, 256, 0, stream>>>(chunk, sidx, xs);
  for (int l = 0; l < LL; ++l) {
    k_rms<<<MCASC, 256, 0, stream>>>(xs, c_n1 + (size_t)l * HH, hbuf, HH);
    k_gemm_bf16<<<grid_g(MCASC, 3 * HH), 256, 0, stream>>>(
        hbuf, wb_cqkv[l], nullptr, nullptr, qkvb, MCASC, 3 * HH, HH);
    k_casc_attn<<<BB * NHH, 256, 0, stream>>>(qkvb, attno);
    k_gemm_bf16<<<grid_g(MCASC, HH), 256, 0, stream>>>(
        attno, wb_co[l], nullptr, xs, xs, MCASC, HH, HH);
    k_rms<<<MCASC, 256, 0, stream>>>(xs, c_n2 + (size_t)l * HH, hbuf, HH);
    k_gemm_swiglu<<<grid_s(MCASC, FFP), 256, 0, stream>>>(
        hbuf, wb_cw1[l], wb_cw3[l], gcas, MCASC, FFP, HH);
    k_gemm_bf16<<<grid_g(MCASC, HH), 256, 0, stream>>>(
        gcas, wb_cw2[l], nullptr, xs, xs, MCASC, HH, FFP);
  }
  k_scatter<<<BB * KK, 256, 0, stream>>>(xs, sidx, proc);

  // ---- ChunkDecoder ----
  k_gemm_bf16<<<grid_g(MDEC, HH), 256, 0, stream>>>(
      token, wb_inw, d_in_b, nullptr, qbuf, MDEC, HH, HH);
  k_gemm_bf16<<<grid_g(MCASC, HH), 256, 0, stream>>>(
      proc, wb_inw + (size_t)HH * HH, d_in_b + HH, nullptr, kbuf, MCASC, HH, HH);
  k_gemm_bf16<<<grid_g(MCASC, HH), 256, 0, stream>>>(
      proc, wb_inw + (size_t)2 * HH * HH, d_in_b + 2 * HH, nullptr, vbuf, MCASC, HH, HH);
  k_dec_attn<<<(BB * NHH * SS) / 8, 256, 0, stream>>>(qbuf, kbuf, vbuf, attn);
  // dec out-proj + bias + token residual -> t1
  k_gemm_bf16<<<grid_g(MDEC, HH), 256, 0, stream>>>(
      attn, wb_outw, d_out_b, token, t1, MDEC, HH, HH);
  // y = rms(t1) -> d_out ; r = rms(y) -> attn (reuse)
  k_rms<<<MDEC, 256, 0, stream>>>(t1, d_nw, out, HH);
  k_rms<<<MDEC, 256, 0, stream>>>(out, d_fnw, attn, HH);
  // gated = silu(r@w1^T) * (r@w3^T)   (pad cols come out exactly 0)
  k_gemm_swiglu<<<grid_s(MDEC, FFP), 256, 0, stream>>>(
      attn, wb_dw1, wb_dw3, gated, MDEC, FFP, HH);
  // out = y + gated @ w2^T   (resid = d_out, per-element RMW)
  k_gemm_bf16<<<grid_g(MDEC, HH), 256, 0, stream>>>(
      gated, wb_dw2, nullptr, out, out, MDEC, HH, FFP);
}
